// RelationalAttentionLayer_9500467659501
// MI455X (gfx1250) — matrix operand order repaired and checked by reading, not compiled, not run on hardware
//
#include <hip/hip_runtime.h>
#include <hip/hip_bf16.h>

typedef __attribute__((ext_vector_type(16))) _Float16 v16h;
typedef __attribute__((ext_vector_type(8)))  float    v8f;

#define B_   2
#define S_   1024
#define H_   12
#define D_   64
#define DM_  768
#define KT_  128   // score columns held in LDS per iteration (96 KB tile)

// ---------------- WMMA helpers ----------------
__device__ __forceinline__ v8f wmma_f16(v16h a, v16h b, v8f c) {
  // D = A(16x32 f16) * B(32x16 f16) + C(16x16 f32)
  return __builtin_amdgcn_wmma_f32_16x16x32_f16(false, a, false, b, (short)0, c,
                                                false, false);
}
__device__ __forceinline__ v8f vzero8() {
  v8f z;
#pragma unroll
  for (int i = 0; i < 8; ++i) z[i] = 0.f;
  return z;
}
__device__ __forceinline__ v16h hzero16() {
  v16h z;
#pragma unroll
  for (int i = 0; i < 16; ++i) z[i] = (_Float16)0.f;
  return z;
}

// ---- Fragment loaders per CDNA5 ISA 7.12.2 (wave32) ----
// A 16(M)x32(K) f16: lanes 0-15 row M=lane, K={0..7,16..23}; lanes 16-31 row M=lane-16, K={8..15,24..31}
__device__ __forceinline__ v16h load_a_f16(const _Float16* A, long lda, int m0, int k0, int lane) {
  int r = lane & 15, hf = lane >> 4;
  const _Float16* p0 = A + (long)(m0 + r) * lda + k0 + 8 * hf;
  v16h a;
#pragma unroll
  for (int j = 0; j < 8; ++j) { a[j] = p0[j]; a[8 + j] = p0[16 + j]; }
  return a;
}
__device__ __forceinline__ v16h load_a_f32(const float* A, long lda, int m0, int k0, int lane) {
  int r = lane & 15, hf = lane >> 4;
  const float* p0 = A + (long)(m0 + r) * lda + k0 + 8 * hf;
  v16h a;
#pragma unroll
  for (int j = 0; j < 8; ++j) { a[j] = (_Float16)p0[j]; a[8 + j] = (_Float16)p0[16 + j]; }
  return a;
}
// B 32(K)x16(N), row-major source B[k][n], leading dim ldb (strided; used only for relation_v fp32)
__device__ __forceinline__ v16h load_b_f32(const float* Bm, long ldb, int k0, int n0, int lane) {
  int c = lane & 15, hf = lane >> 4;
  const float* p = Bm + (long)(k0 + 8 * hf) * ldb + n0 + c;
  v16h v;
#pragma unroll
  for (int j = 0; j < 8; ++j) { v[j] = (_Float16)p[(long)j * ldb]; v[8 + j] = (_Float16)p[(long)(16 + j) * ldb]; }
  return v;
}
// B from transposed storage: element(k,n) = Bt[n*ld + k] (k contiguous -> 32B/lane, coalesced)
__device__ __forceinline__ v16h load_bT_f16(const _Float16* Bt, long ld, int k0, int n0, int lane) {
  int c = lane & 15, hf = lane >> 4;
  const _Float16* p = Bt + (long)(n0 + c) * ld + k0 + 8 * hf;
  v16h v;
#pragma unroll
  for (int j = 0; j < 8; ++j) { v[j] = p[j]; v[8 + j] = p[16 + j]; }
  return v;
}
__device__ __forceinline__ v16h load_bT_f32(const float* Bt, long ld, int k0, int n0, int lane) {
  int c = lane & 15, hf = lane >> 4;
  const float* p = Bt + (long)(n0 + c) * ld + k0 + 8 * hf;
  v16h v;
#pragma unroll
  for (int j = 0; j < 8; ++j) { v[j] = (_Float16)p[j]; v[8 + j] = (_Float16)p[16 + j]; }
  return v;
}

// ---------------- Kernel 0: weight transpose+convert  Wt[n][k] = (f16)W[k][n] ----------------
__global__ __launch_bounds__(256)
void w_to_f16T(const float* __restrict__ W, _Float16* __restrict__ Wt, int K, int N) {
  long t = (long)blockIdx.x * 256 + threadIdx.x;
  if (t >= (long)K * N) return;
  int k = (int)(t % K), n = (int)(t / K);
  Wt[t] = (_Float16)W[(long)k * N + n];   // write coalesced in k
}

// ---------------- Kernel 1/4: WMMA GEMM  C = scale*(A @ W + bias) ----------------
// Templated on A dtype and output mode so the inner loop and epilogue are branch-free.
// AF16: 0 = A fp32, 1 = A f16.  SPLIT: 0 = plain fp32 [M][N]; 1 = f16 [B][H][S][D];
// 2 = f16 transposed [B][H][D][S].
template <int AF16, int SPLIT>
__global__ __launch_bounds__(128)
void wmma_gemm(const void* __restrict__ Ap, const _Float16* __restrict__ Wt,
               const float* __restrict__ bias, float* __restrict__ out32,
               _Float16* __restrict__ out16, float scale, int M, int Kd, int N) {
  int lane = threadIdx.x & 31, wid = threadIdx.x >> 5;
  int n0 = blockIdx.x * 16;
  int m0 = (blockIdx.y * 4 + wid) * 16;
  if (m0 >= M) return;

  v8f acc = vzero8();
  for (int kk = 0; kk < Kd; kk += 32) {
    v16h a;
    if (AF16) a = load_a_f16((const _Float16*)Ap, Kd, m0, kk, lane);
    else      a = load_a_f32((const float*)Ap, Kd, m0, kk, lane);
    v16h b = load_bT_f16(Wt, Kd, kk, n0, lane);   // contiguous 32B/lane
    acc = wmma_f16(a, b, acc);
  }
  int c = lane & 15, hf = lane >> 4;
  int n = n0 + c;
  float bv = bias[n];
#pragma unroll
  for (int r = 0; r < 8; ++r) {
    int m = m0 + r + 8 * hf;                 // C layout: VGPR r -> M=r (lo lanes) / r+8 (hi lanes)
    float v = (acc[r] + bv) * scale;
    if (SPLIT == 1) {
      int bb = m >> 10, ss = m & (S_ - 1);
      int h = n >> 6, d = n & 63;
      out16[((long)(bb * H_ + h) * S_ + ss) * D_ + d] = (_Float16)v;
    } else if (SPLIT == 2) {
      int bb = m >> 10, ss = m & (S_ - 1);
      int h = n >> 6, d = n & 63;
      out16[((long)(bb * H_ + h) * D_ + d) * S_ + ss] = (_Float16)v;
    } else {
      out32[(long)m * N + n] = v;
    }
  }
}

// ---------------- Kernel 2: scores = softmax over k of (qk + q_rel) ----------------
// One workgroup per (b, 16-row q tile); 16 waves. relation_k streamed exactly once.
__global__ __launch_bounds__(512)
void scores_kernel(const _Float16* __restrict__ Qf, const _Float16* __restrict__ Kf,
                   const float* __restrict__ relK, _Float16* __restrict__ Sf) {
  __shared__ float sc[H_ * 16 * KT_];   // 96 KB score tile [h][q][col]
  int lane = threadIdx.x & 31, wid = threadIdx.x >> 5;
  int b = blockIdx.y;
  int q0 = blockIdx.x * 16;
  int c = lane & 15, hf = lane >> 4;

  // Persistent A fragments.
  v16h aQ0 = hzero16(), aQ1 = hzero16();    // qk: wave h owns head h, rows q0..q0+15
  if (wid < H_) {
    const _Float16* Qh = Qf + (long)(b * H_ + wid) * S_ * D_;
    aQ0 = load_a_f16(Qh, D_, q0, 0, lane);
    aQ1 = load_a_f16(Qh, D_, q0, 32, lane);
  }
  // q_rel: wave i owns q-row q0+i, heads stacked in M (rows 12..15 zero)
  v16h aH0 = hzero16(), aH1 = hzero16();
  if (c < H_) {
    const _Float16* p = Qf + ((long)(b * H_ + c) * S_ + (q0 + wid)) * D_;
#pragma unroll
    for (int j = 0; j < 8; ++j) {
      aH0[j]     = p[8 * hf + j];        aH0[8 + j] = p[16 + 8 * hf + j];
      aH1[j]     = p[32 + 8 * hf + j];   aH1[8 + j] = p[48 + 8 * hf + j];
    }
  }
  const float* relQrow = relK + ((long)b * S_ + (q0 + wid)) * (long)S_ * D_;

  for (int kt = 0; kt < S_; kt += KT_) {
    // ---- phase 1: qk (wave h owns head h) ----
    if (wid < H_) {
      const _Float16* Kh = Kf + (long)(b * H_ + wid) * S_ * D_;
      for (int nt = 0; nt < KT_ / 16; ++nt) {
        v8f acc = vzero8();
        acc = wmma_f16(aQ0, load_bT_f16(Kh, D_, 0,  kt + nt * 16, lane), acc);
        acc = wmma_f16(aQ1, load_bT_f16(Kh, D_, 32, kt + nt * 16, lane), acc);
#pragma unroll
        for (int r = 0; r < 8; ++r)
          sc[(wid * 16 + (r + 8 * hf)) * KT_ + nt * 16 + c] = acc[r];
      }
    } else {
      // idle waves warm the relation_k stream for phase 2 (global_prefetch_b8)
      for (int t = lane; t < KT_; t += 32)
        __builtin_prefetch(relQrow + (long)(kt + t) * D_, 0, 1);
    }
    __syncthreads();
    // ---- phase 2: q_rel (wave i owns q-row i; heads stacked in M) ----
    for (int nt = 0; nt < KT_ / 16; ++nt) {
      v8f acc = vzero8();
      acc = wmma_f16(aH0, load_bT_f32(relQrow, D_, 0,  kt + nt * 16, lane), acc);
      acc = wmma_f16(aH1, load_bT_f32(relQrow, D_, 32, kt + nt * 16, lane), acc);
#pragma unroll
      for (int r = 0; r < 8; ++r) {
        int h = r + 8 * hf;
        if (h < H_) sc[(h * 16 + wid) * KT_ + nt * 16 + c] += acc[r];
      }
    }
    __syncthreads();
    // ---- phase 3: spill score tile to global (f16, L2-resident) ----
    for (int t = threadIdx.x; t < H_ * 16 * KT_; t += 512) {
      int col = t % KT_, hq = t / KT_;
      int h = hq >> 4, q = hq & 15;
      Sf[((long)(b * H_ + h) * S_ + (q0 + q)) * S_ + kt + col] = (_Float16)sc[t];
    }
    __syncthreads();
  }

  // ---- softmax in place over k (rows partitioned across waves) ----
  __threadfence();
  for (int row = wid; row < H_ * 16; row += 16) {
    int h = row / 16, q = row % 16;
    _Float16* rp = Sf + ((long)(b * H_ + h) * S_ + (q0 + q)) * S_;
    float m = -3.0e38f;
    for (int j = lane; j < S_; j += 32) m = fmaxf(m, (float)rp[j]);
#pragma unroll
    for (int off = 16; off; off >>= 1) m = fmaxf(m, __shfl_xor(m, off, 32));
    float s = 0.f;
    for (int j = lane; j < S_; j += 32) {
      float e = __expf((float)rp[j] - m);
      s += e;
      rp[j] = (_Float16)e;
    }
#pragma unroll
    for (int off = 16; off; off >>= 1) s += __shfl_xor(s, off, 32);
    float inv = 1.f / s;
    for (int j = lane; j < S_; j += 32) rp[j] = (_Float16)((float)rp[j] * inv);
  }
}

// ---------------- Kernel 3: out = probs@V + probs@relation_v (merged heads, f16) ----------------
// Vt is V transposed per head: [B][H][D][S] so wv B-fragments are contiguous.
__global__ __launch_bounds__(512)
void out_kernel(const _Float16* __restrict__ Sf, const _Float16* __restrict__ Vt,
                const float* __restrict__ relV, _Float16* __restrict__ Af) {
  __shared__ float Oacc[H_ * 16 * D_];   // 48 KB [h][q][d]
  int lane = threadIdx.x & 31, wid = threadIdx.x >> 5;
  int b = blockIdx.y, q0 = blockIdx.x * 16;
  int c = lane & 15, hf = lane >> 4;
  const float* rVrow = relV + ((long)b * S_ + (q0 + wid)) * (long)S_ * D_;

  // ---- phase 1: wv (wave h owns head h) ----
  if (wid < H_) {
    const _Float16* Ph  = Sf + ((long)(b * H_ + wid) * S_ + q0) * S_;
    const _Float16* Vth = Vt + (long)(b * H_ + wid) * D_ * S_;
    v8f acc[4];
#pragma unroll
    for (int dt = 0; dt < 4; ++dt) acc[dt] = vzero8();
    for (int kk = 0; kk < S_; kk += 32) {
      v16h a = load_a_f16(Ph, S_, 0, kk, lane);
#pragma unroll
      for (int dt = 0; dt < 4; ++dt)
        acc[dt] = wmma_f16(a, load_bT_f16(Vth, S_, kk, dt * 16, lane), acc[dt]);
    }
#pragma unroll
    for (int dt = 0; dt < 4; ++dt)
#pragma unroll
      for (int r = 0; r < 8; ++r)
        Oacc[(wid * 16 + (r + 8 * hf)) * D_ + dt * 16 + c] = acc[dt][r];
  } else {
    // idle waves warm the relation_v stream (global_prefetch_b8)
    for (int t = lane; t < 256; t += 32)
      __builtin_prefetch(rVrow + (long)t * D_, 0, 1);
  }
  __syncthreads();

  // ---- phase 2: w_rel (wave i owns q-row i; heads stacked in M) ----
  {
    const _Float16* aBase =
        (c < H_) ? Sf + ((long)(b * H_ + c) * S_ + (q0 + wid)) * S_ : (const _Float16*)0;
    v8f acc[4];
#pragma unroll
    for (int dt = 0; dt < 4; ++dt) acc[dt] = vzero8();
    for (int kk = 0; kk < S_; kk += 32) {
      v16h a = hzero16();
      if (aBase) {
#pragma unroll
        for (int j = 0; j < 8; ++j) {
          a[j]     = aBase[kk + 8 * hf + j];
          a[8 + j] = aBase[kk + 16 + 8 * hf + j];
        }
      }
#pragma unroll
      for (int dt = 0; dt < 4; ++dt)
        acc[dt] = wmma_f16(a, load_b_f32(rVrow, D_, kk, dt * 16, lane), acc[dt]);
    }
#pragma unroll
    for (int dt = 0; dt < 4; ++dt)
#pragma unroll
      for (int r = 0; r < 8; ++r) {
        int h = r + 8 * hf;
        if (h < H_) Oacc[(h * 16 + wid) * D_ + dt * 16 + c] += acc[dt][r];
      }
  }
  __syncthreads();

  // ---- write merged-head attn_out [B*S][768] f16 ----
  for (int t = threadIdx.x; t < H_ * 16 * D_; t += 512) {
    int d = t & 63, hq = t >> 6;
    int q = hq & 15, h = hq >> 4;
    Af[((long)(b * S_) + q0 + q) * DM_ + h * D_ + d] = (_Float16)Oacc[t];
  }
}

// ---------------- launch ----------------
extern "C" void kernel_launch(void* const* d_in, const int* in_sizes, int n_in,
                              void* d_out, int out_size, void* d_ws, size_t ws_size,
                              hipStream_t stream) {
  (void)in_sizes; (void)n_in; (void)out_size; (void)ws_size;
  const float* queries = (const float*)d_in[0];
  const float* keys    = (const float*)d_in[1];
  const float* values  = (const float*)d_in[2];
  const float* relK    = (const float*)d_in[3];
  const float* relV    = (const float*)d_in[4];
  const float* Wq = (const float*)d_in[5];  const float* bq = (const float*)d_in[6];
  const float* Wk = (const float*)d_in[7];  const float* bk = (const float*)d_in[8];
  const float* Wv = (const float*)d_in[9];  const float* bv = (const float*)d_in[10];
  const float* Wo = (const float*)d_in[11]; const float* bo = (const float*)d_in[12];

  char* ws = (char*)d_ws;
  size_t off = 0;
  _Float16* Qf  = (_Float16*)(ws + off); off += (size_t)B_ * H_ * S_ * D_ * 2;   // 3 MB
  _Float16* Kf  = (_Float16*)(ws + off); off += (size_t)B_ * H_ * S_ * D_ * 2;   // 3 MB
  _Float16* Vt  = (_Float16*)(ws + off); off += (size_t)B_ * H_ * S_ * D_ * 2;   // 3 MB
  _Float16* Sf  = (_Float16*)(ws + off); off += (size_t)B_ * H_ * S_ * S_ * 2;   // 48 MB
  _Float16* Af  = (_Float16*)(ws + off); off += (size_t)B_ * S_ * DM_ * 2;       // 3 MB
  _Float16* WqT = (_Float16*)(ws + off); off += (size_t)DM_ * DM_ * 2;           // ~1.1 MB
  _Float16* WkT = (_Float16*)(ws + off); off += (size_t)DM_ * DM_ * 2;
  _Float16* WvT = (_Float16*)(ws + off); off += (size_t)DM_ * DM_ * 2;
  _Float16* WoT = (_Float16*)(ws + off); off += (size_t)DM_ * DM_ * 2;

  // 0) weight transpose/convert (tiny)
  int wElems = DM_ * DM_;
  dim3 tblk(256), tgrd((wElems + 255) / 256);
  w_to_f16T<<<tgrd, tblk, 0, stream>>>(Wq, WqT, DM_, DM_);
  w_to_f16T<<<tgrd, tblk, 0, stream>>>(Wk, WkT, DM_, DM_);
  w_to_f16T<<<tgrd, tblk, 0, stream>>>(Wv, WvT, DM_, DM_);
  w_to_f16T<<<tgrd, tblk, 0, stream>>>(Wo, WoT, DM_, DM_);

  dim3 gblk(128), ggrd(DM_ / 16, (B_ * S_) / 64);
  // Q gets the folded 1/d_key = 1/64 scaling (d_key^-0.5 pre-scale * 1/sqrt(d_key) in scores)
  wmma_gemm<0, 1><<<ggrd, gblk, 0, stream>>>(queries, WqT, bq, nullptr, Qf,
                                             1.0f / 64.0f, B_ * S_, DM_, DM_);
  wmma_gemm<0, 1><<<ggrd, gblk, 0, stream>>>(keys, WkT, bk, nullptr, Kf,
                                             1.0f, B_ * S_, DM_, DM_);
  wmma_gemm<0, 2><<<ggrd, gblk, 0, stream>>>(values, WvT, bv, nullptr, Vt,
                                             1.0f, B_ * S_, DM_, DM_);

  dim3 ablk(512), agrd(S_ / 16, B_);
  scores_kernel<<<agrd, ablk, 0, stream>>>(Qf, Kf, relK, Sf);
  out_kernel<<<agrd, ablk, 0, stream>>>(Sf, Vt, relV, Af);

  wmma_gemm<1, 0><<<ggrd, gblk, 0, stream>>>(Af, WoT, bo, (float*)d_out, nullptr,
                                             1.0f, B_ * S_, DM_, DM_);
}
